// LSHSelfAttention_25761213841558
// MI455X (gfx1250) — compile-verified, matrix operand order checked by reading, not run on hardware
//
#include <hip/hip_runtime.h>
#include <math.h>

// ---------------- types ----------------
typedef __attribute__((ext_vector_type(16))) __bf16    v16bf;
typedef __attribute__((ext_vector_type(8)))  float     v8f;
typedef __attribute__((ext_vector_type(4)))  unsigned  v4u;
typedef __attribute__((ext_vector_type(8)))  int       v8i;
typedef __attribute__((ext_vector_type(4)))  int       v4i;

#define B_      2
#define T_      4096
#define DIM     1024
#define HEADS   16
#define DH      64
#define NH      8
#define NB      64
#define BH      (B_ * HEADS)        // 32 merged batch*heads
#define NCHUNK  (NH * NB)           // 512 global chunks per bh
#define SROWS   (NH * T_)           // 32768 sorted rows per bh
#define MTOT    (B_ * T_)           // 8192 rows for the projections

static __device__ __forceinline__ __bf16 f2bf(float f) {
    unsigned u = __builtin_bit_cast(unsigned, f);
    unsigned r = (u + 0x7FFFu + ((u >> 16) & 1u)) >> 16;
    unsigned short s = (unsigned short)r;
    return __builtin_bit_cast(__bf16, s);
}

// Load a 16x32 bf16 A/B fragment row-slice: rowptr points at element k0 of the
// lane's matrix row; koff = 0 (lanes 0-15) or 8 (lanes 16-31).
static __device__ __forceinline__ v16bf load_row_frag(const __bf16* rowptr, int koff) {
    v16bf f;
#pragma unroll
    for (int i = 0; i < 8; ++i) {
        f[i]     = rowptr[koff + i];
        f[i + 8] = rowptr[16 + koff + i];
    }
    return f;
}

// ---------------- TDM: 1-D async DMA global -> LDS (bf16 elements) ----------------
// Builds a D# (group0 + group1) for a linear tile of `nelem` 2-byte elements and
// issues TENSOR_LOAD_TO_LDS. Tracked by TENSORcnt.
#if __has_builtin(__builtin_amdgcn_tensor_load_to_lds)
#define HAVE_TDM 1
static __device__ __forceinline__ void tdm_load_1d(unsigned lds_off, const void* gptr,
                                                   unsigned nelem) {
    unsigned long long ga = (unsigned long long)(uintptr_t)gptr;
    // group0: count=1 (valid user D#), lds_addr, global_addr[56:0], type=2 ("image")
    v4u g0 = { 1u,
               lds_off,
               (unsigned)ga,
               (unsigned)((ga >> 32) & 0x01FFFFFFu) | (2u << 30) };
    // group1: wg_mask=0, data_size=1 (2B), 1-D tile: tensor_dim0=tile_dim0=nelem
    v8i g1 = { (int)(1u << 16),                       // [17:16] data_size = 2 bytes
               (int)((nelem & 0xFFFFu) << 16),        // [63:48] tensor_dim0 lo
               (int)((nelem >> 16) & 0xFFFFu),        // [79:64] tensor_dim0 hi
               (int)((nelem & 0xFFFFu) << 16),        // [127:112] tile_dim0
               0,                                     // tile_dim1 / tile_dim2 = 0
               (int)nelem,                            // [191:160] tensor_dim0_stride lo
               0, 0 };
    v4i z4 = {};
#if defined(__clang_major__) && __clang_major__ >= 23
    v8i z8 = {};
    __builtin_amdgcn_tensor_load_to_lds(g0, g1, z4, z4, z8, 0);
#else
    __builtin_amdgcn_tensor_load_to_lds(g0, g1, z4, z4, 0);
#endif
}
#else
#define HAVE_TDM 0
#endif

// ---------------- fp32 -> bf16 convert ----------------
__global__ void cvt_bf16_kernel(const float* __restrict__ in, __bf16* __restrict__ out, int n) {
    int i = blockIdx.x * 256 + threadIdx.x;
    if (i < n) out[i] = f2bf(in[i]);
}

// ---------------- WMMA GEMM: C[M,N] = A[M,K] * B[N,K]^T (+bias) ----------------
// Block = 128 threads (4 waves); wave w owns M-tile w; each wave computes a
// 16x64 strip (4 N-tiles) reusing one A fragment per k-step (4x A reuse).
__global__ void gemm_wmma_kernel(const __bf16* __restrict__ A, const __bf16* __restrict__ B,
                                 const float* __restrict__ bias, float* __restrict__ C,
                                 int M, int N, int K) {
    const int lane  = threadIdx.x & 31;
    const int w     = threadIdx.x >> 5;       // 0..3
    const int n0    = blockIdx.x * 64;
    const int m0    = blockIdx.y * 64 + w * 16;
    const int mrow  = lane & 15;
    const int koff  = (lane >> 4) << 3;       // 0 or 8
    const int rbase = koff;                   // C row offset 0 or 8

    const __bf16* arow = A + (size_t)(m0 + mrow) * K;
    v8f acc[4] = {};
    for (int k0 = 0; k0 < K; k0 += 32) {
        v16bf af = load_row_frag(arow + k0, koff);
#pragma unroll
        for (int nt = 0; nt < 4; ++nt) {
            const __bf16* brow = B + (size_t)(n0 + nt * 16 + mrow) * K;
            v16bf bf = load_row_frag(brow + k0, koff);
            acc[nt] = __builtin_amdgcn_wmma_f32_16x16x32_bf16(false, af, false, bf,
                                                              (short)0, acc[nt], false, false);
        }
    }
#pragma unroll
    for (int nt = 0; nt < 4; ++nt) {
        const int col = n0 + nt * 16 + (lane & 15);
        const float bv = bias ? bias[col] : 0.0f;
#pragma unroll
        for (int r = 0; r < 8; ++r)
            C[(size_t)(m0 + r + rbase) * N + col] = acc[nt][r] + bv;
    }
}

// ---------------- LSH hashing: bucket = argmax([r, -r]) ----------------
// grid = (T/256, NH, BH), block = 256
__global__ void hash_kernel(const float* __restrict__ qk, const float* __restrict__ rot,
                            int* __restrict__ buckets) {
    __shared__ float rotLds[DH * 32];
    const int h  = blockIdx.y;
    const int bh = blockIdx.z;
    const int b = bh >> 4, head = bh & 15;
    for (int i = threadIdx.x; i < DH * 32; i += 256) {
        int f = i >> 5, ri = i & 31;
        rotLds[i] = rot[f * (NH * 32) + h * 32 + ri];   // rotations[f][h][i]
    }
    __syncthreads();
    const int t = blockIdx.x * 256 + threadIdx.x;
    const float* q = qk + ((size_t)b * T_ + t) * DIM + head * DH;
    float r[32];
#pragma unroll
    for (int i = 0; i < 32; ++i) r[i] = 0.0f;
    for (int f = 0; f < DH; ++f) {
        float qf = q[f];
#pragma unroll
        for (int i = 0; i < 32; ++i) r[i] += qf * rotLds[f * 32 + i];
    }
    float best = -3.4e38f; int bi = 0;
#pragma unroll
    for (int j = 0; j < 64; ++j) {
        float v = (j < 32) ? r[j] : -r[j - 32];
        if (v > best) { best = v; bi = j; }
    }
    buckets[((size_t)bh * NH + h) * T_ + t] = bi;
}

// ---------------- stable counting sort per (bh, round) ----------------
// grid = (NH, BH), block = 256; each thread owns 16 consecutive tokens (stable).
__global__ void sort_kernel(const int* __restrict__ buckets, int* __restrict__ st) {
    __shared__ unsigned short hist[64 * 256];   // [bin][thread], 32KB
    __shared__ unsigned partials[256];
    const int h = blockIdx.x, bh = blockIdx.y;
    const int tid = threadIdx.x;
    const int* bk = buckets + ((size_t)bh * NH + h) * T_;

    for (int i = tid; i < 64 * 256; i += 256) hist[i] = 0;
    __syncthreads();

    const int t0 = tid * 16;
    int mybkt[16];
#pragma unroll
    for (int j = 0; j < 16; ++j) {
        int bb = bk[t0 + j];
        mybkt[j] = bb;
        hist[bb * 256 + tid]++;          // slot unique per thread
    }
    __syncthreads();

    unsigned running = 0;
    for (int j = 0; j < 64; ++j) {
        int f = tid * 64 + j;
        unsigned tmp = hist[f];
        hist[f] = (unsigned short)running;
        running += tmp;
    }
    partials[tid] = running;
    __syncthreads();
    if (tid == 0) {
        unsigned run = 0;
        for (int i = 0; i < 256; ++i) { unsigned tmp = partials[i]; partials[i] = run; run += tmp; }
    }
    __syncthreads();
    {
        unsigned base = partials[tid];
        for (int j = 0; j < 64; ++j)
            hist[tid * 64 + j] = (unsigned short)(hist[tid * 64 + j] + base);
    }
    __syncthreads();

    int* out = st + ((size_t)bh * NH + h) * T_;
#pragma unroll
    for (int j = 0; j < 16; ++j) {
        int bb = mybkt[j];
        unsigned pos = hist[bb * 256 + tid]++;
        out[pos] = t0 + j;
    }
}

// ---------------- gather sorted Q / normalized K / V (bf16) ----------------
__global__ void gather_kernel(const float* __restrict__ qk, const float* __restrict__ v,
                              const int* __restrict__ st,
                              __bf16* __restrict__ sq, __bf16* __restrict__ sk,
                              __bf16* __restrict__ sv) {
    const size_t row = (size_t)blockIdx.x * 256 + threadIdx.x;  // [bh][NH*T]
    const int bh = (int)(row >> 15);
    const int b = bh >> 4, head = bh & 15;
    const int t = st[row];
    const float* qs = qk + ((size_t)b * T_ + t) * DIM + head * DH;
    const float* vs = v  + ((size_t)b * T_ + t) * DIM + head * DH;
    float qv[DH], nrm = 0.0f;
#pragma unroll
    for (int f = 0; f < DH; ++f) { qv[f] = qs[f]; nrm += qv[f] * qv[f]; }
    float rn = 1.0f / fmaxf(sqrtf(nrm), 1e-12f);
#pragma unroll
    for (int f = 0; f < DH; ++f) {
        sq[row * DH + f] = f2bf(qv[f]);
        sk[row * DH + f] = f2bf(qv[f] * rn);
        sv[row * DH + f] = f2bf(vs[f]);
    }
}

// ---------------- chunk-local attention ----------------
// grid = (NCHUNK, BH), block = 256 (8 waves).
// LDS: dots 64x128 fp32 (32KB) + V tile 128x64 bf16 (16KB).
// V tile is DMA'd in by the Tensor Data Mover (wave 0) and overlapped with the
// Q.K^T WMMA stage; completion via s_wait_tensorcnt before the block barrier.
__global__ void attn_kernel(const __bf16* __restrict__ sq, const __bf16* __restrict__ sk,
                            const __bf16* __restrict__ sv, const int* __restrict__ st,
                            float* __restrict__ o, float* __restrict__ logits) {
    __shared__ float  dots[64 * 128];
    __shared__ __bf16 Vs[128 * DH];
    const int bh  = blockIdx.y;
    const int cg  = blockIdx.x;                 // global chunk: h*64 + c
    const int h   = cg >> 6;
    const int cgp = (cg + NCHUNK - 1) & (NCHUNK - 1);   // look-one-back (rolls across h)
    const int tid = threadIdx.x, lane = tid & 31, w = tid >> 5;
    const size_t sbase = (size_t)bh * SROWS;
    const int mrow  = lane & 15;
    const int koff  = (lane >> 4) << 3;
    const int rbase = (lane >> 4) << 3;

#if HAVE_TDM
    // ----- kick off async V-tile DMA (two linear 64x64 bf16 half-chunks)
    if (w == 0) {
        unsigned vs_off = (unsigned)(uintptr_t)(void*)Vs;   // LDS byte offset (low 32 bits)
        tdm_load_1d(vs_off,        sv + (sbase + (size_t)cg  * 64) * DH, 64 * DH);
        tdm_load_1d(vs_off + 8192, sv + (sbase + (size_t)cgp * 64) * DH, 64 * DH);
    }
#else
    for (int i = tid; i < 128 * DH; i += 256) {
        int j = i >> 6, c = i & 63;
        size_t gr = sbase + ((j < 64) ? (size_t)cg * 64 + j : (size_t)cgp * 64 + (j - 64));
        Vs[i] = sv[gr * DH + c];
    }
#endif

    // ----- dots = Q(64x64) . K^T(64x128); wave w owns N columns [w*16, w*16+16)
    {
        const int n = w * 16 + mrow;   // 0..127 key index
        const size_t krow = sbase + ((n < 64) ? (size_t)cg * 64 + n
                                              : (size_t)cgp * 64 + (n - 64));
        const __bf16* kr = sk + krow * DH;
        v8f acc[4] = {};
        for (int k0 = 0; k0 < DH; k0 += 32) {
            v16bf bf = load_row_frag(kr + k0, koff);
#pragma unroll
            for (int m = 0; m < 4; ++m) {
                const __bf16* qr = sq + (sbase + (size_t)cg * 64 + m * 16 + mrow) * DH;
                v16bf af = load_row_frag(qr + k0, koff);
                acc[m] = __builtin_amdgcn_wmma_f32_16x16x32_bf16(false, af, false, bf,
                                                                 (short)0, acc[m], false, false);
            }
        }
#pragma unroll
        for (int m = 0; m < 4; ++m)
#pragma unroll
            for (int r = 0; r < 8; ++r)
                dots[(m * 16 + r + rbase) * 128 + w * 16 + (lane & 15)] = acc[m][r] * 0.125f;
    }
#if HAVE_TDM
    if (w == 0) __builtin_amdgcn_s_wait_tensorcnt(0);   // V tile resident before barrier
#endif
    __syncthreads();

    // ----- self-mask + softmax (one row per thread, threads 0..63)
    if (tid < 64) {
        const int r = tid;
        const int idq = st[sbase + (size_t)cg * 64 + r];
        float mx = -3.4e38f;
        for (int j = 0; j < 128; ++j) {
            int idk = st[sbase + ((j < 64) ? (size_t)cg * 64 + j
                                           : (size_t)cgp * 64 + (j - 64))];
            float d = dots[r * 128 + j];
            if (idq == idk) d = -50000.0f;
            dots[r * 128 + j] = d;
            mx = fmaxf(mx, d);
        }
        float s = 0.0f;
        for (int j = 0; j < 128; ++j) s += expf(dots[r * 128 + j] - mx);
        float lse = mx + logf(s);
        for (int j = 0; j < 128; ++j) dots[r * 128 + j] = expf(dots[r * 128 + j] - lse);
        logits[((size_t)bh * NH + h) * T_ + idq] = lse;   // scatter to unsorted order
    }
    __syncthreads();

    // ----- O = P(64x128) . V(128x64); wave w: mtile = w>>1, two 16-wide N tiles
    {
        const int mtile = w >> 1;
        v8f acc2[2] = {};
        for (int k0 = 0; k0 < 128; k0 += 32) {
            const float* pr = &dots[(mtile * 16 + mrow) * 128 + k0];
            v16bf af;
#pragma unroll
            for (int i = 0; i < 8; ++i) {
                af[i]     = f2bf(pr[koff + i]);
                af[i + 8] = f2bf(pr[16 + koff + i]);
            }
#pragma unroll
            for (int nt = 0; nt < 2; ++nt) {
                const int n = ((w & 1) * 2 + nt) * 16 + mrow;   // dh column 0..63
                v16bf bf;
#pragma unroll
                for (int i = 0; i < 8; ++i) {
                    bf[i]     = Vs[(k0 + koff + i) * DH + n];
                    bf[i + 8] = Vs[(k0 + 16 + koff + i) * DH + n];
                }
                acc2[nt] = __builtin_amdgcn_wmma_f32_16x16x32_bf16(false, af, false, bf,
                                                                   (short)0, acc2[nt], false, false);
            }
        }
#pragma unroll
        for (int nt = 0; nt < 2; ++nt) {
            const int col = ((w & 1) * 2 + nt) * 16 + (lane & 15);
#pragma unroll
            for (int r = 0; r < 8; ++r) {
                const int row = mtile * 16 + r + rbase;
                const int t_orig = st[sbase + (size_t)cg * 64 + row];
                o[(((size_t)bh * NH + h) * T_ + t_orig) * DH + col] = acc2[nt][r];
            }
        }
    }
}

// ---------------- combine hash rounds (softmax over lse) ----------------
__global__ void combine_kernel(const float* __restrict__ o, const float* __restrict__ logits,
                               __bf16* __restrict__ attn) {
    const int idx = blockIdx.x * 256 + threadIdx.x;    // bh*T + t
    const int bh = idx >> 12, t = idx & (T_ - 1);
    const int b = bh >> 4, head = bh & 15;
    float lg[NH], mx = -3.4e38f;
#pragma unroll
    for (int h = 0; h < NH; ++h) {
        lg[h] = logits[((size_t)bh * NH + h) * T_ + t];
        mx = fmaxf(mx, lg[h]);
    }
    float s = 0.0f;
#pragma unroll
    for (int h = 0; h < NH; ++h) { lg[h] = expf(lg[h] - mx); s += lg[h]; }
    const float inv = 1.0f / s;
    __bf16* dst = attn + ((size_t)b * T_ + t) * DIM + head * DH;
    for (int d = 0; d < DH; ++d) {
        float a = 0.0f;
#pragma unroll
        for (int h = 0; h < NH; ++h)
            a += lg[h] * inv * o[(((size_t)bh * NH + h) * T_ + t) * DH + d];
        dst[d] = f2bf(a);
    }
}

// ---------------- host launcher ----------------
extern "C" void kernel_launch(void* const* d_in, const int* in_sizes, int n_in,
                              void* d_out, int out_size, void* d_ws, size_t ws_size,
                              hipStream_t stream) {
    const float* x     = (const float*)d_in[0];
    const float* w_qk  = (const float*)d_in[1];
    const float* w_v   = (const float*)d_in[2];
    const float* w_out = (const float*)d_in[3];
    const float* b_out = (const float*)d_in[4];
    const float* rot   = (const float*)d_in[5];
    float* out = (float*)d_out;

    size_t off = 0;
    char* base = (char*)d_ws;
    auto carve = [&](size_t bytes) -> void* {
        void* p = base + off;
        off += (bytes + 255) & ~(size_t)255;
        return p;
    };
    float*  qk_f32   = (float*)carve((size_t)MTOT * DIM * 4);
    float*  v_f32    = (float*)carve((size_t)MTOT * DIM * 4);
    __bf16* x_bf     = (__bf16*)carve((size_t)MTOT * DIM * 2);
    __bf16* wqk_bf   = (__bf16*)carve((size_t)DIM * DIM * 2);
    __bf16* wv_bf    = (__bf16*)carve((size_t)DIM * DIM * 2);
    __bf16* wout_bf  = (__bf16*)carve((size_t)DIM * DIM * 2);
    int*    buckets  = (int*)carve((size_t)BH * NH * T_ * 4);
    int*    st       = (int*)carve((size_t)BH * NH * T_ * 4);
    __bf16* sq       = (__bf16*)carve((size_t)BH * SROWS * DH * 2);
    __bf16* sk       = (__bf16*)carve((size_t)BH * SROWS * DH * 2);
    __bf16* sv       = (__bf16*)carve((size_t)BH * SROWS * DH * 2);
    float*  o_buf    = (float*)carve((size_t)BH * SROWS * DH * 4);
    float*  logits   = (float*)carve((size_t)BH * SROWS * 4);
    __bf16* attn_bf  = (__bf16*)carve((size_t)MTOT * DIM * 2);
    (void)ws_size; (void)in_sizes; (void)n_in; (void)out_size;

    // 1) fp32 -> bf16 conversions
    {
        int n = MTOT * DIM;
        cvt_bf16_kernel<<<(n + 255) / 256, 256, 0, stream>>>(x, x_bf, n);
        int m = DIM * DIM;
        cvt_bf16_kernel<<<(m + 255) / 256, 256, 0, stream>>>(w_qk, wqk_bf, m);
        cvt_bf16_kernel<<<(m + 255) / 256, 256, 0, stream>>>(w_v, wv_bf, m);
        cvt_bf16_kernel<<<(m + 255) / 256, 256, 0, stream>>>(w_out, wout_bf, m);
    }
    // 2) input projections (WMMA, 16x64 strip per wave)
    {
        dim3 grid(DIM / 64, MTOT / 64);
        gemm_wmma_kernel<<<grid, 128, 0, stream>>>(x_bf, wqk_bf, nullptr, qk_f32, MTOT, DIM, DIM);
        gemm_wmma_kernel<<<grid, 128, 0, stream>>>(x_bf, wv_bf,  nullptr, v_f32,  MTOT, DIM, DIM);
    }
    // 3) LSH hashing
    hash_kernel<<<dim3(T_ / 256, NH, BH), 256, 0, stream>>>(qk_f32, rot, buckets);
    // 4) stable counting sort per (bh, round)
    sort_kernel<<<dim3(NH, BH), 256, 0, stream>>>(buckets, st);
    // 5) gather sorted Q/K/V (bf16)
    gather_kernel<<<(BH * SROWS) / 256, 256, 0, stream>>>(qk_f32, v_f32, st, sq, sk, sv);
    // 6) chunk-local attention (WMMA + TDM async V staging)
    attn_kernel<<<dim3(NCHUNK, BH), 256, 0, stream>>>(sq, sk, sv, st, o_buf, logits);
    // 7) combine hash rounds
    combine_kernel<<<(BH * T_) / 256, 256, 0, stream>>>(o_buf, logits, attn_bf);
    // 8) output projection with bias (WMMA)
    {
        dim3 grid(DIM / 64, MTOT / 64);
        gemm_wmma_kernel<<<grid, 128, 0, stream>>>(attn_bf, wout_bf, b_out, out, MTOT, DIM, DIM);
    }
}
